// FusedTransition_69879117906305
// MI455X (gfx1250) — matrix operand order repaired and checked by reading, not compiled, not run on hardware
//
#include <hip/hip_runtime.h>

typedef __attribute__((ext_vector_type(16))) _Float16 v16h;
typedef __attribute__((ext_vector_type(8)))  _Float16 h8;
typedef __attribute__((ext_vector_type(4)))  _Float16 h4;
typedef __attribute__((ext_vector_type(8)))  float    v8f;

#define CIN   512
#define COUT  256
#define HH    56
#define WW    56
#define HO    28
#define WO    28
#define BN_EPS 1e-5f

// ---------------- prologue: conv weights f32 -> f16 into workspace ----------
__global__ __launch_bounds__(256) void prep_weights(const float* __restrict__ wgt,
                                                    _Float16* __restrict__ wsA) {
  int i = blockIdx.x * 256 + threadIdx.x;       // 32768 threads, 4 elems each
  float4 v = *(const float4*)&wgt[(size_t)i * 4];
  h4 o;
  o[0] = (_Float16)v.x; o[1] = (_Float16)v.y;
  o[2] = (_Float16)v.z; o[3] = (_Float16)v.w;
  *(h4*)&wsA[(size_t)i * 4] = o;
}

// ---------------- main fused kernel ----------------------------------------
// Workgroup: 256 threads (8 wave32). Computes full Cout=256 for a 2x16 pixel
// tile so x is streamed from HBM exactly once (~231 MB total -> ~10us at
// 23.3 TB/s). A-tiles (f16 weights in ws) are staged via
// global_load_async_to_lds_b128 (ASYNCcnt DMA), triple-buffered, one barrier
// per K-step; B-tiles get BN+ReLU+f16 in VALU. GEMM: v_wmma_f32_16x16x32_f16.
__launch_bounds__(256)
__global__ void fused_bn_relu_conv1x1_pool(const float* __restrict__ x,
                                           const float* __restrict__ bn_w,
                                           const float* __restrict__ bn_b,
                                           const float* __restrict__ rmean,
                                           const float* __restrict__ rvar,
                                           const _Float16* __restrict__ wsA,
                                           float* __restrict__ out) {
  __shared__ float s_scale[CIN];
  __shared__ float s_shift[CIN];
  // A tiles: 3 x (256 rows x 32 K), rows padded to 40 f16 (80B, bank-friendly)
  __shared__ __align__(16) _Float16 sA[3][COUT][40];
  // B tiles: 3 x (32 pixels x 32 K), [pixel][k] so fragments are contiguous
  __shared__ __align__(16) _Float16 sB[3][32][40];
  // Pool buffer aliases sA (only live after the K-loop): 256 x 33 f32
  float (*sP)[33] = reinterpret_cast<float(*)[33]>(&sA[0][0][0]);

  const int tid    = threadIdx.x;
  const int lane   = tid & 31;
  const int wv     = tid >> 5;
  const int nsub   = wv & 1;        // pooled-pair row (2h / 2h+1)
  const int mquad  = wv >> 1;       // 64 Cout rows per wave
  const int hi     = lane >> 4;
  const int lane15 = lane & 15;

  const int w0t = blockIdx.x;       // 0..3 (16-wide w tile; tile 3 ragged)
  const int h   = blockIdx.y;       // 0..27
  const int b   = blockIdx.z;       // batch

  // ---- BN coefficients ----
  for (int c = tid; c < CIN; c += 256) {
    float inv = rsqrtf(rvar[c] + BN_EPS);
    float sc  = bn_w[c] * inv;
    s_scale[c] = sc;
    s_shift[c] = bn_b[c] - rmean[c] * sc;
  }

  v8f acc[4] = {};

  // ---- async A-staging: 4 b128 copies per thread per tile ----
  const int ar0 = tid >> 2;                    // base row 0..63
  const int aq  = tid & 3;                     // 16B quarter of the 64B slice
  const uint32_t aGlobBase = (uint32_t)ar0 * (CIN * 2) + (uint32_t)aq * 16;
  const uint32_t ldsA_my   = (uint32_t)(uintptr_t)&sA[0][ar0][aq * 8];
  const uint32_t BUFA      = (uint32_t)sizeof(_Float16) * COUT * 40;  // 20480

  // ---- B-staging: thread owns (pixel p, 4 consecutive channels) ----
  const int p    = tid & 31;
  const int ccol = (tid >> 5) * 4;             // channel offset in tile
  const int prow = p >> 4;
  const int pw   = p & 15;
  const int gh   = 2 * h + prow;
  int gw         = w0t * 16 + pw;
  if (gw > WW - 1) gw = WW - 1;                // clamp: ragged tile cols are
                                               // computed but never stored
  const float* xbase = x + (size_t)b * CIN * HH * WW + (size_t)gh * WW + gw;

  // ---- per-thread fragment base offsets (in f16 elements) ----
  const _Float16* bB_my = &sB[0][nsub * 16 + lane15][8 * hi];
  const _Float16* bA_my = &sA[0][mquad * 64 + lane15][8 * hi];

#define STAGE_A(tile, buf)                                                   \
  do {                                                                       \
    uint32_t gofs = aGlobBase + (uint32_t)(tile) * 64u;                      \
    uint32_t lbase = ldsA_my + (uint32_t)(buf) * BUFA;                       \
    _Pragma("unroll")                                                        \
    for (int ii = 0; ii < 4; ++ii) {                                         \
      uint32_t lds = lbase + (uint32_t)(ii * 64 * 80);                       \
      uint32_t go  = gofs + (uint32_t)(64 * ii) * (CIN * 2);                 \
      asm volatile("global_load_async_to_lds_b128 %0, %1, %2"                \
                   :: "v"(lds), "v"(go), "s"(wsA) : "memory");               \
    }                                                                        \
  } while (0)

#define STAGE_B(tile, buf)                                                   \
  do {                                                                       \
    float bv[4];                                                             \
    _Pragma("unroll")                                                        \
    for (int ii = 0; ii < 4; ++ii)                                           \
      bv[ii] = xbase[(size_t)((tile) * 32 + ccol + ii) * (HH * WW)];         \
    h4 bh;                                                                   \
    _Pragma("unroll")                                                        \
    for (int ii = 0; ii < 4; ++ii) {                                         \
      int c = (tile) * 32 + ccol + ii;                                       \
      bh[ii] = (_Float16)fmaxf(bv[ii] * s_scale[c] + s_shift[c], 0.0f);      \
    }                                                                        \
    *(h4*)&sB[buf][p][ccol] = bh;                                            \
  } while (0)

  // Prime tile 0
  STAGE_A(0, 0);
  STAGE_B(0, 0);

  int cur = 0;
  for (int kk = 0; kk < 16; ++kk) {
    if (kk < 15) {
      int nxt = cur + 1; if (nxt == 3) nxt = 0;
      STAGE_A(kk + 1, nxt);
      STAGE_B(kk + 1, nxt);
      // my 4 copies for tile kk are done once only tile kk+1's 4 remain
      asm volatile("s_wait_asynccnt 0x4" ::: "memory");
    } else {
      asm volatile("s_wait_asynccnt 0x0" ::: "memory");
    }
    __syncthreads();   // all threads' tile-kk staging (async + ds) visible

    // ---- issue ALL fragment DS loads first ----
    const _Float16* pB = bB_my + cur * (32 * 40);
    const _Float16* pA = bA_my + cur * (COUT * 40);
    h8 b0 = *(const h8*)pB;
    h8 b1 = *(const h8*)(pB + 16);
    h8 a0[4], a1[4];
#pragma unroll
    for (int j = 0; j < 4; ++j) {
      a0[j] = *(const h8*)(pA + j * (16 * 40));        // K = 8*hi + 0..7
      a1[j] = *(const h8*)(pA + j * (16 * 40) + 16);   // K = 16 + 8*hi + 0..7
    }
    // scheduling fence: keep the loads above the WMMA group so the backend
    // staggers s_wait_dscnt against WMMA execution instead of lockstepping
    asm volatile("" ::: "memory");

    v16h bf;
#pragma unroll
    for (int e = 0; e < 8; ++e) { bf[e] = b0[e]; bf[e + 8] = b1[e]; }
#pragma unroll
    for (int j = 0; j < 4; ++j) {
      v16h af;
#pragma unroll
      for (int e = 0; e < 8; ++e) { af[e] = a0[j][e]; af[e + 8] = a1[j][e]; }
      acc[j] = __builtin_amdgcn_wmma_f32_16x16x32_f16(
          false, af, false, bf, (short)0, acc[j], false, false);
    }

    cur = cur + 1; if (cur == 3) cur = 0;
  }

  __syncthreads();
  // ---- spill D tiles (VGPR i holds M = i + 8*hi, N = lane15) into sP ----
#pragma unroll
  for (int j = 0; j < 4; ++j) {
#pragma unroll
    for (int i = 0; i < 8; ++i) {
      sP[mquad * 64 + j * 16 + 8 * hi + i][nsub * 16 + lane15] = acc[j][i];
    }
  }
  __syncthreads();

  // ---- 2x2 average pool + store ----
  for (int t = tid; t < COUT * 8; t += 256) {
    int m  = t >> 3;
    int wo = t & 7;
    float v = 0.25f * (sP[m][2 * wo] + sP[m][2 * wo + 1] +
                       sP[m][16 + 2 * wo] + sP[m][16 + 2 * wo + 1]);
    int gwo = w0t * 8 + wo;
    if (gwo < WO) {
      out[(((size_t)b * COUT + m) * HO + h) * WO + gwo] = v;
    }
  }
#undef STAGE_A
#undef STAGE_B
}

extern "C" void kernel_launch(void* const* d_in, const int* in_sizes, int n_in,
                              void* d_out, int out_size, void* d_ws, size_t ws_size,
                              hipStream_t stream) {
  const float* x     = (const float*)d_in[0];
  const float* bn_w  = (const float*)d_in[1];
  const float* bn_b  = (const float*)d_in[2];
  const float* rmean = (const float*)d_in[3];
  const float* rvar  = (const float*)d_in[4];
  const float* wgt   = (const float*)d_in[5];
  float* out         = (float*)d_out;
  _Float16* wsA      = (_Float16*)d_ws;        // 256x512 f16 = 256 KB

  prep_weights<<<dim3(128), dim3(256), 0, stream>>>(wgt, wsA);

  dim3 grid(4, HO, 32);   // w-tiles x pooled rows x batch
  dim3 block(256);
  fused_bn_relu_conv1x1_pool<<<grid, block, 0, stream>>>(
      x, bn_w, bn_b, rmean, rvar, wsA, out);
}
